// MLA_4956392259705
// MI455X (gfx1250) — matrix operand order repaired and checked by reading, not compiled, not run on hardware
//
#include <hip/hip_runtime.h>
#include <hip/hip_bf16.h>

typedef __attribute__((ext_vector_type(16))) __bf16 bf16x16;
typedef __attribute__((ext_vector_type(8)))  __bf16 bf16x8;
typedef __attribute__((ext_vector_type(8)))  float  f32x8;

#define S_LEN 2048
#define NHEAD 16
#define DQK   192
#define DVV   128

// ---------------------------------------------------------------------------
// Elementwise conversion kernels
// ---------------------------------------------------------------------------
__global__ void k_f32_to_bf16(const float* __restrict__ in, __bf16* __restrict__ out, size_t n) {
    size_t i = (size_t)blockIdx.x * blockDim.x + threadIdx.x;
    if (i < n) out[i] = (__bf16)in[i];
}

// W [K][N] fp32 -> Wt [N][K] bf16   (compile-time dims -> no integer divides)
template <int K, int N>
__global__ void k_transpose_bf16(const float* __restrict__ in, __bf16* __restrict__ out) {
    size_t i = (size_t)blockIdx.x * blockDim.x + threadIdx.x;
    if (i < (size_t)K * N) {
        int k = (int)(i / N);
        int n = (int)(i % N);
        out[(size_t)n * K + k] = (__bf16)in[i];
    }
}

// ---------------------------------------------------------------------------
// RMS norm: one block per row, fp32 in, bf16 out
// ---------------------------------------------------------------------------
__global__ __launch_bounds__(256) void k_rmsnorm_bf16(const float* __restrict__ X,
                                                      const float* __restrict__ W,
                                                      __bf16* __restrict__ Y, int C) {
    int row = blockIdx.x;
    const float* x = X + (size_t)row * C;
    __shared__ float red[256];
    float ss = 0.f;
    for (int i = threadIdx.x; i < C; i += 256) { float v = x[i]; ss += v * v; }
    red[threadIdx.x] = ss;
    __syncthreads();
    for (int s = 128; s > 0; s >>= 1) {
        if (threadIdx.x < s) red[threadIdx.x] += red[threadIdx.x + s];
        __syncthreads();
    }
    float inv = rsqrtf(red[0] / (float)C + 1e-5f);
    for (int i = threadIdx.x; i < C; i += 256)
        Y[(size_t)row * C + i] = (__bf16)(x[i] * inv * W[i]);
}

// ---------------------------------------------------------------------------
// bf16 WMMA GEMM:  C[M][N] (fp32) = A[M][K] (bf16, row-major) * Bt[N][K]^T
// Block tile 128x128, BK=64, double-buffered LDS (64KB), 256 threads (8 waves),
// wave tile 64x32 (4x2 accumulators), 2 WMMA K-steps per LDS buffer.
// ---------------------------------------------------------------------------
#define BM 128
#define BN 128
#define BKK 64

template <int N, int K>
__global__ __launch_bounds__(256) void k_gemm_bf16(const __bf16* __restrict__ A,
                                                   const __bf16* __restrict__ Bt,
                                                   float* __restrict__ C) {
    __shared__ __align__(16) __bf16 As[2][BM * BKK];   // 2 x 16 KB
    __shared__ __align__(16) __bf16 Bs[2][BN * BKK];   // 2 x 16 KB

    const int tid  = threadIdx.x;
    const int lane = tid & 31;
    const int wave = tid >> 5;
    const int wm   = wave & 1;   // row group: wm*64
    const int wn   = wave >> 1;  // col group: wn*32
    const int bm   = blockIdx.y * BM;
    const int bn   = blockIdx.x * BN;
    const int half = lane >> 4;
    const int l16  = lane & 15;

    const int lrow = tid >> 3;
    const int lcol = (tid & 7) * 8;
    const __bf16* aptr = A  + (size_t)(bm + lrow) * K + lcol;
    const __bf16* bptr = Bt + (size_t)(bn + lrow) * K + lcol;

    f32x8 acc[4][2];
#pragma unroll
    for (int i = 0; i < 4; ++i)
#pragma unroll
        for (int j = 0; j < 2; ++j)
            acc[i][j] = (f32x8){0.f,0.f,0.f,0.f,0.f,0.f,0.f,0.f};

    bf16x8 areg[4], breg[4];

#pragma unroll
    for (int i = 0; i < 4; ++i) {
        areg[i] = *(const bf16x8*)(aptr + (size_t)(i * 32) * K);
        breg[i] = *(const bf16x8*)(bptr + (size_t)(i * 32) * K);
    }
#pragma unroll
    for (int i = 0; i < 4; ++i) {
        *(bf16x8*)&As[0][(lrow + i * 32) * BKK + lcol] = areg[i];
        *(bf16x8*)&Bs[0][(lrow + i * 32) * BKK + lcol] = breg[i];
    }
    __syncthreads();

    constexpr int T = K / BKK;
    for (int t = 0; t < T; ++t) {
        const int cur = t & 1;
        if (t + 1 < T) {
            const __bf16* ap = aptr + (size_t)(t + 1) * BKK;
            const __bf16* bp = bptr + (size_t)(t + 1) * BKK;
#pragma unroll
            for (int i = 0; i < 4; ++i) {
                areg[i] = *(const bf16x8*)(ap + (size_t)(i * 32) * K);
                breg[i] = *(const bf16x8*)(bp + (size_t)(i * 32) * K);
            }
            if (t + 2 < T) {   // pull t+2 toward cache (global_prefetch_b8)
                __builtin_prefetch(ap + BKK, 0, 0);
                __builtin_prefetch(bp + BKK, 0, 0);
            }
        }

#pragma unroll
        for (int ks = 0; ks < 2; ++ks) {
            bf16x16 afr[4], bfr[2];
#pragma unroll
            for (int mt = 0; mt < 4; ++mt) {
                const __bf16* p = &As[cur][(wm * 64 + mt * 16 + l16) * BKK + ks * 32];
                ((bf16x8*)&afr[mt])[0] = *(const bf16x8*)&p[half * 8];
                ((bf16x8*)&afr[mt])[1] = *(const bf16x8*)&p[16 + half * 8];
            }
#pragma unroll
            for (int nt = 0; nt < 2; ++nt) {
                const __bf16* p = &Bs[cur][(wn * 32 + nt * 16 + l16) * BKK + ks * 32 + half * 8];
                ((bf16x8*)&bfr[nt])[0] = *(const bf16x8*)&p[0];
                ((bf16x8*)&bfr[nt])[1] = *(const bf16x8*)&p[16];
            }
#pragma unroll
            for (int mt = 0; mt < 4; ++mt)
#pragma unroll
                for (int nt = 0; nt < 2; ++nt)
                    acc[mt][nt] = __builtin_amdgcn_wmma_f32_16x16x32_bf16(
                        false, afr[mt], false, bfr[nt], (short)0, acc[mt][nt], false, false);
        }

        if (t + 1 < T) {
#pragma unroll
            for (int i = 0; i < 4; ++i) {
                *(bf16x8*)&As[cur ^ 1][(lrow + i * 32) * BKK + lcol] = areg[i];
                *(bf16x8*)&Bs[cur ^ 1][(lrow + i * 32) * BKK + lcol] = breg[i];
            }
        }
        __syncthreads();
    }

#pragma unroll
    for (int mt = 0; mt < 4; ++mt)
#pragma unroll
        for (int nt = 0; nt < 2; ++nt) {
            const int col  = bn + wn * 32 + nt * 16 + l16;
            const int rowb = bm + wm * 64 + mt * 16 + half * 8;
#pragma unroll
            for (int r = 0; r < 8; ++r)
                C[(size_t)(rowb + r) * N + col] = acc[mt][nt][r];
        }
}

// ---------------------------------------------------------------------------
// RoPE on q/k tail dims + repack fp32 [S][NH*192] -> bf16 head-major [NH][S][192]
// ---------------------------------------------------------------------------
__global__ void k_rope_pack(const float* __restrict__ q, const float* __restrict__ k,
                            __bf16* __restrict__ q_hm, __bf16* __restrict__ k_hm) {
    size_t idx = (size_t)blockIdx.x * blockDim.x + threadIdx.x;
    size_t total = (size_t)S_LEN * NHEAD * DQK;
    if (idx >= total) return;
    int d = (int)(idx % DQK);
    int h = (int)((idx / DQK) % NHEAD);
    int s = (int)(idx / ((size_t)DQK * NHEAD));
    size_t src = (size_t)s * (NHEAD * DQK) + (size_t)h * DQK + d;
    float qv = q[src], kv = k[src];
    if (d >= 128) {
        int j = d - 128;                          // 0..63 within rope dims
        // theta^(-(j%32)/32) = exp2(-(j%32) * log2(theta)/32), log2(1e5)=16.609640474
        float freq = (float)s * exp2f(-(float)(j & 31) * (16.609640474f / 32.0f));
        float c = __cosf(freq), sn = __sinf(freq);
        int jr = (j < 32) ? (j + 32) : (j - 32);
        float sgn = (j < 32) ? -1.f : 1.f;
        size_t srcr = (size_t)s * (NHEAD * DQK) + (size_t)h * DQK + 128 + jr;
        qv = qv * c + sgn * q[srcr] * sn;
        kv = kv * c + sgn * k[srcr] * sn;
    }
    size_t dst = ((size_t)h * S_LEN + s) * DQK + d;
    q_hm[dst] = (__bf16)qv;
    k_hm[dst] = (__bf16)kv;
}

// v fp32 [S][NH*128] -> bf16 TRANSPOSED head-major [NH][DVV][S]
// (so the attention kernel's Vt tile fill is plain coalesced b128 loads)
__global__ void k_v_pack_t(const float* __restrict__ v, __bf16* __restrict__ v_t) {
    size_t idx = (size_t)blockIdx.x * blockDim.x + threadIdx.x;
    size_t total = (size_t)S_LEN * NHEAD * DVV;
    if (idx >= total) return;
    int d = (int)(idx % DVV);
    int h = (int)((idx / DVV) % NHEAD);
    int s = (int)(idx / ((size_t)DVV * NHEAD));
    v_t[((size_t)h * DVV + d) * S_LEN + s] =
        (__bf16)v[(size_t)s * (NHEAD * DVV) + (size_t)h * DVV + d];
}

// ---------------------------------------------------------------------------
// Flash attention (causal): one block per (head, 64-query block), 256 threads.
// V arrives pre-transposed [NH][DVV][S]; all tile fills are vector b128 loads.
// ---------------------------------------------------------------------------
#define QB 64
#define KB 32

__global__ __launch_bounds__(256) void k_mla_attn(const __bf16* __restrict__ Qh,
                                                  const __bf16* __restrict__ Kh,
                                                  const __bf16* __restrict__ Vt_g, /*[NH][DVV][S]*/
                                                  __bf16* __restrict__ Out /*[S][NH*DVV]*/) {
    int h  = blockIdx.y;
    int qb = blockIdx.x;
    int q0 = qb * QB;
    int tid  = threadIdx.x;
    int lane = tid & 31;
    int wave = tid >> 5;
    int half = lane >> 4;
    int l16  = lane & 15;

    __shared__ __align__(16) __bf16 Qs[QB * DQK];   // 24 KB
    __shared__ __align__(16) __bf16 Ks[KB * DQK];   // 12 KB
    __shared__ __align__(16) __bf16 Vt[DVV * KB];   //  8 KB ([dv][key])
    __shared__ __align__(16) float  Sc[QB * KB];    //  8 KB
    __shared__ __align__(16) __bf16 Ps[QB * KB];    //  4 KB
    __shared__ float mrow[QB], lrow[QB], arow[QB];

    const __bf16* qsrc = Qh + ((size_t)h * S_LEN + q0) * DQK;
#pragma unroll
    for (int i = 0; i < (QB * DQK / 8) / 256; ++i) {
        int c = tid + i * 256;
        *(bf16x8*)&Qs[c * 8] = *(const bf16x8*)&qsrc[(size_t)c * 8];
    }
    if (tid < QB) { mrow[tid] = -1e30f; lrow[tid] = 0.f; }

    f32x8 o[4];
#pragma unroll
    for (int i = 0; i < 4; ++i) o[i] = (f32x8){0.f,0.f,0.f,0.f,0.f,0.f,0.f,0.f};

    const int rt = wave & 3;   // O row tile (0..3) -> rows rt*16
    const int cg = wave >> 2;  // O col group (0..1) -> cols cg*64 .. +63

    const float scale = 0.072168784f; // 1/sqrt(192)
    const int kmax = q0 + QB;
    const __bf16* vbase = Vt_g + (size_t)h * DVV * S_LEN;

    for (int k0 = 0; k0 < kmax; k0 += KB) {
        __syncthreads();   // prev iteration fully consumed Ks/Vt/Ps
        const __bf16* ksrc = Kh + ((size_t)h * S_LEN + k0) * DQK;
#pragma unroll
        for (int i = 0; i < (KB * DQK / 8) / 256; ++i) {
            int c = tid + i * 256;
            *(bf16x8*)&Ks[c * 8] = *(const bf16x8*)&ksrc[(size_t)c * 8];
        }
        // Vt tile [128][32]: 512 chunks of 8, 2 per thread, coalesced rows
#pragma unroll
        for (int i = 0; i < (DVV * KB / 8) / 256; ++i) {
            int c  = tid + i * 256;
            int dv = c >> 2;            // 4 chunks per 32-wide row
            int cl = (c & 3) * 8;
            *(bf16x8*)&Vt[dv * KB + cl] =
                *(const bf16x8*)&vbase[(size_t)dv * S_LEN + k0 + cl];
        }
        __syncthreads();

        // ---- scores: 4x2 grid of 16x16 tiles, one tile per wave ----
        {
            const int mS = wave & 3;
            const int nS = wave >> 2;
            f32x8 sa = (f32x8){0.f,0.f,0.f,0.f,0.f,0.f,0.f,0.f};
#pragma unroll
            for (int kk = 0; kk < DQK; kk += 32) {
                bf16x16 af, bfv;
                const __bf16* pa = &Qs[(mS * 16 + l16) * DQK + kk];
                ((bf16x8*)&af)[0] = *(const bf16x8*)&pa[half * 8];
                ((bf16x8*)&af)[1] = *(const bf16x8*)&pa[16 + half * 8];
                const __bf16* pb = &Ks[(nS * 16 + l16) * DQK + kk + half * 8];
                ((bf16x8*)&bfv)[0] = *(const bf16x8*)&pb[0];
                ((bf16x8*)&bfv)[1] = *(const bf16x8*)&pb[16];
                sa = __builtin_amdgcn_wmma_f32_16x16x32_bf16(
                    false, af, false, bfv, (short)0, sa, false, false);
            }
#pragma unroll
            for (int r = 0; r < 8; ++r)
                Sc[(mS * 16 + half * 8 + r) * KB + nS * 16 + l16] = sa[r] * scale;
        }
        __syncthreads();

        // ---- per-row online softmax (threads 0..63) ----
        if (tid < QB) {
            int qg = q0 + tid;
            float mold = mrow[tid];
            float mb = -1e30f;
            float sv[KB];
#pragma unroll
            for (int c = 0; c < KB; ++c) {
                float s = (k0 + c <= qg) ? Sc[tid * KB + c] : -1e30f;
                sv[c] = s;
                mb = fmaxf(mb, s);
            }
            float mnew  = fmaxf(mold, mb);
            float alpha = __expf(mold - mnew);
            float sum = 0.f;
#pragma unroll
            for (int c = 0; c < KB; ++c) {
                float p = __expf(sv[c] - mnew);
                sum += p;
                Ps[tid * KB + c] = (__bf16)p;
            }
            mrow[tid] = mnew;
            lrow[tid] = lrow[tid] * alpha + sum;
            arow[tid] = alpha;
        }
        __syncthreads();

        // ---- rescale O and accumulate P x V ----
        {
            bf16x16 pf;
            const __bf16* pp = &Ps[(rt * 16 + l16) * KB];
            ((bf16x8*)&pf)[0] = *(const bf16x8*)&pp[half * 8];
            ((bf16x8*)&pf)[1] = *(const bf16x8*)&pp[16 + half * 8];
            float al[8];
#pragma unroll
            for (int r = 0; r < 8; ++r) al[r] = arow[rt * 16 + half * 8 + r];
#pragma unroll
            for (int ct = 0; ct < 4; ++ct) {
#pragma unroll
                for (int r = 0; r < 8; ++r)
                    o[ct][r] *= al[r];
                bf16x16 vf;
                const __bf16* pv = &Vt[(cg * 64 + ct * 16 + l16) * KB + half * 8];
                ((bf16x8*)&vf)[0] = *(const bf16x8*)&pv[0];
                ((bf16x8*)&vf)[1] = *(const bf16x8*)&pv[16];
                o[ct] = __builtin_amdgcn_wmma_f32_16x16x32_bf16(
                    false, pf, false, vf, (short)0, o[ct], false, false);
            }
        }
    }
    __syncthreads();

    // ---- finalize: multiply by 1/l, write bf16 [S][NH*DVV] ----
    float linv[8];
#pragma unroll
    for (int r = 0; r < 8; ++r) linv[r] = 1.0f / lrow[rt * 16 + half * 8 + r];
#pragma unroll
    for (int ct = 0; ct < 4; ++ct) {
        const int colg = h * DVV + cg * 64 + ct * 16 + l16;
#pragma unroll
        for (int r = 0; r < 8; ++r) {
            const int qrow = rt * 16 + half * 8 + r;
            Out[(size_t)(q0 + qrow) * (NHEAD * DVV) + colg] = (__bf16)(o[ct][r] * linv[r]);
        }
    }
}

// ---------------------------------------------------------------------------
// Host-side launch
// ---------------------------------------------------------------------------
extern "C" void kernel_launch(void* const* d_in, const int* in_sizes, int n_in,
                              void* d_out, int out_size, void* d_ws, size_t ws_size,
                              hipStream_t stream) {
    const float* x        = (const float*)d_in[0];  // [2048][2048]
    const float* wq_down  = (const float*)d_in[1];  // [2048][1536]
    const float* q_norm_w = (const float*)d_in[2];  // [1536]
    const float* wq_up    = (const float*)d_in[3];  // [1536][3072]
    const float* wkv_down = (const float*)d_in[4];  // [2048][512]
    const float* kv_norm  = (const float*)d_in[5];  // [512]
    const float* wk_up    = (const float*)d_in[6];  // [512][3072]
    const float* wv_up    = (const float*)d_in[7];  // [512][2048]
    const float* wo       = (const float*)d_in[8];  // [2048][2048]
    float* out = (float*)d_out;

    char* ws = (char*)d_ws;
    size_t off = 0;
    auto alloc = [&](size_t bytes) -> void* {
        void* p = ws + off;
        off += (bytes + 255) & ~(size_t)255;
        return p;
    };

    const int H = 2048, QR = 1536, KVR = 512, QKH = NHEAD * DQK, VH = NHEAD * DVV;

    __bf16* x_bf       = (__bf16*)alloc((size_t)S_LEN * H * 2);
    __bf16* wq_down_t  = (__bf16*)alloc((size_t)QR * H * 2);
    __bf16* wq_up_t    = (__bf16*)alloc((size_t)QKH * QR * 2);
    __bf16* wkv_down_t = (__bf16*)alloc((size_t)KVR * H * 2);
    __bf16* wk_up_t    = (__bf16*)alloc((size_t)QKH * KVR * 2);
    __bf16* wv_up_t    = (__bf16*)alloc((size_t)VH * KVR * 2);
    __bf16* wo_t       = (__bf16*)alloc((size_t)H * VH * 2);
    float*  q_lat_raw  = (float*)alloc((size_t)S_LEN * QR * 4);
    __bf16* q_lat_bf   = (__bf16*)alloc((size_t)S_LEN * QR * 2);
    float*  kv_raw     = (float*)alloc((size_t)S_LEN * KVR * 4);
    __bf16* kv_bf      = (__bf16*)alloc((size_t)S_LEN * KVR * 2);
    float*  q_f32      = (float*)alloc((size_t)S_LEN * QKH * 4);
    float*  k_f32      = (float*)alloc((size_t)S_LEN * QKH * 4);
    float*  v_f32      = (float*)alloc((size_t)S_LEN * VH * 4);
    __bf16* q_hm       = (__bf16*)alloc((size_t)NHEAD * S_LEN * DQK * 2);
    __bf16* k_hm       = (__bf16*)alloc((size_t)NHEAD * S_LEN * DQK * 2);
    __bf16* v_t        = (__bf16*)alloc((size_t)NHEAD * DVV * S_LEN * 2);
    __bf16* attn_bf    = (__bf16*)alloc((size_t)S_LEN * VH * 2);

    auto cdiv = [](size_t a, size_t b) { return (a + b - 1) / b; };

    // --- conversions ---
    {
        size_t n = (size_t)S_LEN * H;
        k_f32_to_bf16<<<cdiv(n, 256), 256, 0, stream>>>(x, x_bf, n);
    }
    k_transpose_bf16<2048, 1536><<<cdiv((size_t)H * QR, 256), 256, 0, stream>>>(wq_down, wq_down_t);
    k_transpose_bf16<1536, 3072><<<cdiv((size_t)QR * QKH, 256), 256, 0, stream>>>(wq_up, wq_up_t);
    k_transpose_bf16<2048, 512><<<cdiv((size_t)H * KVR, 256), 256, 0, stream>>>(wkv_down, wkv_down_t);
    k_transpose_bf16<512, 3072><<<cdiv((size_t)KVR * QKH, 256), 256, 0, stream>>>(wk_up, wk_up_t);
    k_transpose_bf16<512, 2048><<<cdiv((size_t)KVR * VH, 256), 256, 0, stream>>>(wv_up, wv_up_t);
    k_transpose_bf16<2048, 2048><<<cdiv((size_t)VH * H, 256), 256, 0, stream>>>(wo, wo_t);

    // --- q path ---
    k_gemm_bf16<1536, 2048><<<dim3(QR / BN, S_LEN / BM), 256, 0, stream>>>(x_bf, wq_down_t, q_lat_raw);
    k_rmsnorm_bf16<<<S_LEN, 256, 0, stream>>>(q_lat_raw, q_norm_w, q_lat_bf, QR);
    k_gemm_bf16<3072, 1536><<<dim3(QKH / BN, S_LEN / BM), 256, 0, stream>>>(q_lat_bf, wq_up_t, q_f32);

    // --- kv path ---
    k_gemm_bf16<512, 2048><<<dim3(KVR / BN, S_LEN / BM), 256, 0, stream>>>(x_bf, wkv_down_t, kv_raw);
    k_rmsnorm_bf16<<<S_LEN, 256, 0, stream>>>(kv_raw, kv_norm, kv_bf, KVR);
    k_gemm_bf16<3072, 512><<<dim3(QKH / BN, S_LEN / BM), 256, 0, stream>>>(kv_bf, wk_up_t, k_f32);
    k_gemm_bf16<2048, 512><<<dim3(VH / BN, S_LEN / BM), 256, 0, stream>>>(kv_bf, wv_up_t, v_f32);

    // --- rope + repack ---
    {
        size_t n = (size_t)S_LEN * NHEAD * DQK;
        k_rope_pack<<<cdiv(n, 256), 256, 0, stream>>>(q_f32, k_f32, q_hm, k_hm);
    }
    {
        size_t n = (size_t)S_LEN * NHEAD * DVV;
        k_v_pack_t<<<cdiv(n, 256), 256, 0, stream>>>(v_f32, v_t);
    }

    // --- attention ---
    k_mla_attn<<<dim3(S_LEN / QB, NHEAD), 256, 0, stream>>>(q_hm, k_hm, v_t, attn_bf);

    // --- output projection ---
    k_gemm_bf16<2048, 2048><<<dim3(H / BN, S_LEN / BM), 256, 0, stream>>>(attn_bf, wo_t, out);
}
